// GraphLayer_34875134444200
// MI455X (gfx1250) — matrix operand order repaired and checked by reading, not compiled, not run on hardware
//
#include <hip/hip_runtime.h>

typedef __attribute__((ext_vector_type(16))) __bf16 v16bf;
typedef __attribute__((ext_vector_type(8)))  __bf16 v8bf;
typedef __attribute__((ext_vector_type(8)))  float  v8f;

static constexpr int Bsz = 8;
static constexpr int Nn  = 2048;
static constexpr int Dd  = 512;

static __device__ __forceinline__ v16bf cat8(v8bf lo, v8bf hi) {
  return __builtin_shufflevector(lo, hi, 0,1,2,3,4,5,6,7,8,9,10,11,12,13,14,15);
}

static __device__ __forceinline__ v8f wmma_bf16(v16bf a, v16bf b, v8f c) {
  return __builtin_amdgcn_wmma_f32_16x16x32_bf16(false, a, false, b, (short)0, c,
                                                 false, false);
}

// ---------------------------------------------------------------------------
// Kernel 0: fp32 -> bf16 conversion of x and the three weight matrices.
// ---------------------------------------------------------------------------
__global__ __launch_bounds__(256) void cvt_bf16(
    const float* __restrict__ x,
    const float* __restrict__ wq, const float* __restrict__ wk, const float* __restrict__ wv,
    __bf16* __restrict__ xb,
    __bf16* __restrict__ wqb, __bf16* __restrict__ wkb, __bf16* __restrict__ wvb)
{
  const int stride = gridDim.x * blockDim.x;
  const int gid = blockIdx.x * blockDim.x + threadIdx.x;
  for (int i = gid; i < Bsz * Nn * Dd; i += stride) xb[i] = (__bf16)x[i];
  for (int i = gid; i < Dd * Dd; i += stride) {
    wqb[i] = (__bf16)wq[i];
    wkb[i] = (__bf16)wk[i];
    wvb[i] = (__bf16)wv[i];
  }
}

// ---------------------------------------------------------------------------
// Kernel 1: fused QKV projection.  Y = X @ W^T + b  (X: [B*N, D], W: [D, D]).
// Block = 8 waves, tile 128 rows x 64 cols; each wave: 16 rows x 64 cols x 3.
// Wk/Wv tiles are addressed as immediate offsets from the Wq pointer
// (ws layout is contiguous), and operand registers are software-pipelined so
// the next tile's loads issue before the current WMMA's wait.
// ---------------------------------------------------------------------------
__global__ __launch_bounds__(256) void qkv_proj(
    const __bf16* __restrict__ Xb,
    const __bf16* __restrict__ Wqb,            // Wkb = Wqb+WN, Wvb = Wqb+2*WN
    const float* __restrict__ bq, const float* __restrict__ bk, const float* __restrict__ bv,
    __bf16* __restrict__ Qb, __bf16* __restrict__ Kb, __bf16* __restrict__ Vt)
{
  constexpr int WN = Dd * Dd;                  // 262144 elems = 512 KB offset
  const int wave = threadIdx.x >> 5;
  const int lane = threadIdx.x & 31;
  const int m    = lane & 15;
  const int half = lane >> 4;

  const int row0 = blockIdx.x * 128 + wave * 16;  // flat row in [0, B*N)
  const int ob   = blockIdx.y * 64;               // output-feature base

  v8f aq[4] = {}, ak[4] = {}, av[4] = {};

  const __bf16* xr = Xb + (size_t)(row0 + m) * Dd + half * 16;
  const __bf16* wp[4];
#pragma unroll
  for (int t = 0; t < 4; ++t)
    wp[t] = Wqb + (size_t)(ob + t * 16 + m) * Dd + half * 16;

  v16bf A  = cat8(*(const v8bf*)(xr), *(const v8bf*)(xr + 8));
  v16bf Bq = *(const v16bf*)(wp[0]);
  v16bf Bk = *(const v16bf*)(wp[0] + WN);
  v16bf Bv = *(const v16bf*)(wp[0] + 2 * WN);

  for (int d = 0; d < Dd; d += 32) {
#pragma unroll
    for (int t = 0; t < 4; ++t) {
      v16bf An = A, Bqn = Bq, Bkn = Bk, Bvn = Bv;
      const int tn = (t + 1) & 3;
      const int dn = (t == 3) ? d + 32 : d;
      if (dn < Dd) {                            // prefetch next operand set
        Bqn = *(const v16bf*)(wp[tn] + dn);
        Bkn = *(const v16bf*)(wp[tn] + dn + WN);
        Bvn = *(const v16bf*)(wp[tn] + dn + 2 * WN);
        if (t == 3)
          An = cat8(*(const v8bf*)(xr + dn), *(const v8bf*)(xr + dn + 8));
      }
      aq[t] = wmma_bf16(A, Bq, aq[t]);
      ak[t] = wmma_bf16(A, Bk, ak[t]);
      av[t] = wmma_bf16(A, Bv, av[t]);
      A = An; Bq = Bqn; Bk = Bkn; Bv = Bvn;
    }
  }

  const int bidx = row0 / Nn;
  const int nloc = (row0 % Nn) + half * 8;
#pragma unroll
  for (int t = 0; t < 4; ++t) {
    const int col = ob + t * 16 + m;
    const float biasq = bq[col], biask = bk[col], biasv = bv[col];
    v8bf vpack;
#pragma unroll
    for (int r = 0; r < 8; ++r) {
      const int grow = row0 + half * 8 + r;     // C layout: row = r + half*8
      Qb[(size_t)grow * Dd + col] = (__bf16)(aq[t][r] + biasq);
      Kb[(size_t)grow * Dd + col] = (__bf16)(ak[t][r] + biask);
      vpack[r] = (__bf16)(av[t][r] + biasv);
    }
    // transposed V store: rows are consecutive -> one 16B store per tile
    *(v8bf*)(Vt + (size_t)bidx * Dd * Nn + (size_t)col * Nn + nloc) = vpack;
  }
}

// ---------------------------------------------------------------------------
// Kernel 2: single-pass graph attention.
// Block = 16 waves (512 thr) owns 64 queries of one batch. Key-blocks of 64:
//   S phase : wave(w) computes S[16q x 16k] over D=512 (pipelined WMMA chain)
//   w = adj * exp(S); row-sums -> den via shuffle-reduce + ds_add_f32
//   P -> LDS (bf16), barrier, PV phase: wave owns 16q x 128d (pipelined)
// Final: out = accO / den.
// ---------------------------------------------------------------------------
__global__ __launch_bounds__(512) void graph_attn(
    const __bf16* __restrict__ Qb, const __bf16* __restrict__ Kb,
    const __bf16* __restrict__ Vt,
    const float* __restrict__ adj, float* __restrict__ out)
{
  __shared__ __bf16 Pl[64 * 72];   // P tile, row stride 72 to stagger banks
  __shared__ float  den[64];

  const int wave = threadIdx.x >> 5;
  const int lane = threadIdx.x & 31;
  const int m    = lane & 15;
  const int half = lane >> 4;

  const int b  = blockIdx.y;
  const int qb = blockIdx.x * 64;

  const int qs = (wave & 3) * 16;    // q sub-tile (S and PV phases)
  const int ks = (wave >> 2) * 16;   // S phase: key sub-tile
  const int dq = (wave >> 2) * 128;  // PV phase: D quarter

  if (threadIdx.x < 64) den[threadIdx.x] = 0.0f;
  __syncthreads();

  v8f accO[8] = {};

  // loop-invariant per-lane row pointers (tile addresses become immediates)
  const __bf16* qp   = Qb + (size_t)(b * Nn + qb + qs + m) * Dd + half * 16;
  const __bf16* vrow = Vt + (size_t)b * Dd * Nn + (size_t)(dq + m) * Nn + half * 16;
  const float*  arow = adj + (size_t)b * Nn * Nn
                           + (size_t)(qb + qs + half * 8) * Nn + ks + m;

  for (int kb0 = 0; kb0 < Nn; kb0 += 64) {
    // ---- S = Q K^T (16q x 16k per wave, pipelined over D) ----
    const __bf16* kp = Kb + (size_t)(b * Nn + kb0 + ks + m) * Dd + half * 16;
    v16bf A  = cat8(*(const v8bf*)qp, *(const v8bf*)(qp + 8));
    v16bf Bk = *(const v16bf*)kp;
    v8f s = {};
    for (int d = 0; d < Dd; d += 32) {
      v16bf An = A, Bn = Bk;
      if (d + 32 < Dd) {
        An = cat8(*(const v8bf*)(qp + d + 32), *(const v8bf*)(qp + d + 40));
        Bn = *(const v16bf*)(kp + d + 32);
      }
      s = wmma_bf16(A, Bk, s);
      A = An; Bk = Bn;
    }

    // hint the next key-block's adjacency stream into cache
    if (kb0 + 64 < Nn) __builtin_prefetch((const void*)(arow + kb0 + 64), 0, 3);

    // ---- w = adj * exp(S); stage P (bf16) into LDS ----
#pragma unroll
    for (int r = 0; r < 8; ++r) {
      const float w = arow[kb0 + r * Nn] * __expf(s[r]);
      s[r] = w;
      Pl[(qs + half * 8 + r) * 72 + (ks + m)] = (__bf16)w;
    }

    // ---- row sums: butterfly across the 16 columns, then LDS atomic ----
#pragma unroll
    for (int r = 0; r < 8; ++r) {
      float v = s[r];
      v += __shfl_xor(v, 1, 16);
      v += __shfl_xor(v, 2, 16);
      v += __shfl_xor(v, 4, 16);
      v += __shfl_xor(v, 8, 16);
      if (m == 0) atomicAdd(&den[qs + half * 8 + r], v);
    }
    __syncthreads();

    // ---- O += P @ V  (A from LDS in A-layout, B = Vt rows, pipelined) ----
    const __bf16* vp = vrow + kb0;
#pragma unroll
    for (int kc = 0; kc < 2; ++kc) {
      const __bf16* pr = &Pl[(qs + m) * 72 + kc * 32 + half * 16];
      v16bf A  = cat8(*(const v8bf*)pr, *(const v8bf*)(pr + 8));
      v16bf Bv = *(const v16bf*)(vp + kc * 32);
#pragma unroll
      for (int dt = 0; dt < 8; ++dt) {
        v16bf Bn = Bv;
        if (dt < 7)
          Bn = *(const v16bf*)(vp + (dt + 1) * 16 * Nn + kc * 32);
        accO[dt] = wmma_bf16(A, Bv, accO[dt]);
        Bv = Bn;
      }
    }
    __syncthreads();  // P/LDS reused next key-block
  }

  // ---- normalize and store fp32 ----
#pragma unroll
  for (int dt = 0; dt < 8; ++dt) {
    const int dcol = dq + dt * 16 + m;
#pragma unroll
    for (int r = 0; r < 8; ++r) {
      const int q = qb + qs + half * 8 + r;
      out[(size_t)b * Nn * Dd + (size_t)q * Dd + dcol] =
          accO[dt][r] / den[qs + half * 8 + r];
    }
  }
}

// ---------------------------------------------------------------------------
extern "C" void kernel_launch(void* const* d_in, const int* in_sizes, int n_in,
                              void* d_out, int out_size, void* d_ws, size_t ws_size,
                              hipStream_t stream) {
  const float* x    = (const float*)d_in[0];
  const float* adj  = (const float*)d_in[1];
  const float* Wq_w = (const float*)d_in[2];
  const float* Wq_b = (const float*)d_in[3];
  const float* Wk_w = (const float*)d_in[4];
  const float* Wk_b = (const float*)d_in[5];
  const float* Wv_w = (const float*)d_in[6];
  const float* Wv_b = (const float*)d_in[7];
  float* out = (float*)d_out;

  // Workspace layout (bf16 elements):
  //   Qb[B*N*D] | Kb[B*N*D] | Vt[B*D*N] | Xb[B*N*D] | Wqb | Wkb | Wvb
  //   (Wqb/Wkb/Wvb contiguity is relied upon by qkv_proj's immediate offsets)
  __bf16* wsb = (__bf16*)d_ws;
  const size_t XN = (size_t)Bsz * Nn * Dd;   // 8,388,608
  const size_t WN = (size_t)Dd * Dd;         //   262,144
  __bf16* Qb  = wsb;
  __bf16* Kb  = Qb + XN;
  __bf16* Vt  = Kb + XN;
  __bf16* Xb  = Vt + XN;
  __bf16* Wqb = Xb + XN;
  __bf16* Wkb = Wqb + WN;
  __bf16* Wvb = Wkb + WN;

  cvt_bf16<<<2048, 256, 0, stream>>>(x, Wq_w, Wk_w, Wv_w, Xb, Wqb, Wkb, Wvb);

  dim3 gproj(Bsz * Nn / 128, Dd / 64);       // (128, 8)
  qkv_proj<<<gproj, 256, 0, stream>>>(Xb, Wqb,
                                      Wq_b, Wk_b, Wv_b, Qb, Kb, Vt);

  dim3 gattn(Nn / 64, Bsz);                  // (32, 8)
  graph_attn<<<gattn, 512, 0, stream>>>(Qb, Kb, Vt, adj, out);
}
